// TTrain_73229192396884
// MI455X (gfx1250) — compile-verified, matrix operand order checked by reading, not run on hardware
//
#include <hip/hip_runtime.h>

// TT (MPS) log-likelihood, B=256 sequences, T=1024, d=4 symbols, bond D=32.
// Per-sequence prefix scan over 32x32 core matrices:
//   - 1 block per sequence, 8 waves per block, each wave owns a 128-step chunk.
//   - Each step: P <- P @ core[x_t] = 4x v_wmma_f32_16x16x32_f16 (C from inline 0),
//     B operands = 4 cores pinned in VGPRs, symbol select via wave-uniform scalar
//     branch (EXEC stays all-ones as WMMA requires).
//   - Inf-norm rescale each step; wave-wide max via pure-VALU DPP butterfly
//     (quad_perm xor1/xor2, row_half_mirror, row_mirror, permlanex16) - no LDS
//     round trips or dscnt waits in the reduction.
//   - UNSCALED C is stored to LDS right after the WMMAs so the store+load
//     relayout overlaps the max/log/rcp chain; the 1/Z scale is applied when
//     rebuilding A (f32->packed f16). The chunk left in LDS is off by its last
//     Z, compensated exactly via sAcc = acc - log(Z_last).
//   - After __syncthreads(), wave 0 folds left boundary through the 8 chunk
//     matrices and applies the right boundary.

typedef __attribute__((ext_vector_type(16))) _Float16 v16h;
typedef __attribute__((ext_vector_type(8)))  float    v8f;
typedef __attribute__((ext_vector_type(2)))  _Float16 h2v;

#define TT_T   1024
#define TT_NW  8      // waves per block
#define TT_CH  128    // steps per wave = TT_T / TT_NW
#define TT_STR 34     // padded LDS row stride in floats (32 + 2)

// cvt_pkrtz returns a vector of __fp16; WMMA builtins take _Float16 vectors.
__device__ __forceinline__ h2v tt_pack(float a, float b) {
  return __builtin_bit_cast(h2v, __builtin_amdgcn_cvt_pkrtz(a, b));
}

// Wave32 all-lanes max, pure VALU (DPP + permlanex16): no LDS, no waits.
__device__ __forceinline__ float tt_wave_max(float x) {
  int v;
  v = __builtin_bit_cast(int, x);   // xor 1 within quads
  x = fmaxf(x, __builtin_bit_cast(float,
        __builtin_amdgcn_update_dpp(v, v, 0xB1, 0xF, 0xF, false)));
  v = __builtin_bit_cast(int, x);   // xor 2 within quads
  x = fmaxf(x, __builtin_bit_cast(float,
        __builtin_amdgcn_update_dpp(v, v, 0x4E, 0xF, 0xF, false)));
  v = __builtin_bit_cast(int, x);   // row_half_mirror: combines quad pairs
  x = fmaxf(x, __builtin_bit_cast(float,
        __builtin_amdgcn_update_dpp(v, v, 0x141, 0xF, 0xF, false)));
  v = __builtin_bit_cast(int, x);   // row_mirror: combines 8-groups within 16
  x = fmaxf(x, __builtin_bit_cast(float,
        __builtin_amdgcn_update_dpp(v, v, 0x140, 0xF, 0xF, false)));
  v = __builtin_bit_cast(int, x);   // swap lane halves 0-15 <-> 16-31
  x = fmaxf(x, __builtin_bit_cast(float,
        __builtin_amdgcn_permlanex16(v, v, 0x76543210, 0xFEDCBA98, false, false)));
  return x;
}

__device__ __forceinline__ float tt_wave_sum(float x) {
  int v;
  v = __builtin_bit_cast(int, x);
  x = x + __builtin_bit_cast(float,
        __builtin_amdgcn_update_dpp(v, v, 0xB1, 0xF, 0xF, false));
  v = __builtin_bit_cast(int, x);
  x = x + __builtin_bit_cast(float,
        __builtin_amdgcn_update_dpp(v, v, 0x4E, 0xF, 0xF, false));
  v = __builtin_bit_cast(int, x);
  x = x + __builtin_bit_cast(float,
        __builtin_amdgcn_update_dpp(v, v, 0x141, 0xF, 0xF, false));
  v = __builtin_bit_cast(int, x);
  x = x + __builtin_bit_cast(float,
        __builtin_amdgcn_update_dpp(v, v, 0x140, 0xF, 0xF, false));
  v = __builtin_bit_cast(int, x);
  x = x + __builtin_bit_cast(float,
        __builtin_amdgcn_permlanex16(v, v, 0x76543210, 0xFEDCBA98, false, false));
  return x;
}

__global__ __launch_bounds__(256)
void tt_logprob_kernel(const int* __restrict__ X,
                       const float* __restrict__ core,
                       const float* __restrict__ lb,
                       const float* __restrict__ rb,
                       float* __restrict__ out) {
  __shared__ float sP[TT_NW][32 * TT_STR];
  __shared__ float sAcc[TT_NW];

  const int b    = blockIdx.x;
  const int tid  = threadIdx.x;
  const int wid  = tid >> 5;
  const int lane = tid & 31;
  const bool lo  = lane < 16;
  const int col  = lane & 15;
  const int koff = lo ? 0 : 16;     // K offset for this half-wave (A/B f16 layout)

  // ---- Load the 4 cores as WMMA B-tiles (32x16 f16), 2 column-tiles each ----
  // B k order used here (contiguous, same as A): lanes 0-15: N=lane, VGPR r holds K=2r,2r+1;
  //                           lanes 16-31: N=lane-16, K=2r+16,2r+17.
  v16h Bt[4][2];
#pragma unroll
  for (int x = 0; x < 4; ++x) {
#pragma unroll
    for (int tj = 0; tj < 2; ++tj) {
#pragma unroll
      for (int r = 0; r < 8; ++r) {
        float e0 = core[x * 1024 + (2 * r     + koff) * 32 + tj * 16 + col];
        float e1 = core[x * 1024 + (2 * r + 1 + koff) * 32 + tj * 16 + col];
        h2v p = tt_pack(e0, e1);
        Bt[x][tj][2 * r]     = p[0];
        Bt[x][tj][2 * r + 1] = p[1];
      }
    }
  }

  // ---- This wave's 128 symbols: 4 per lane, one b128 load ----
  const int t0 = wid * TT_CH;
  const int4 xv = *(const int4*)(X + b * TT_T + t0 + lane * 4);

  // ---- A = Identity (contiguous k order, as B: lanes 0-15 M=lane K=i, lanes 16-31 K=i+16) ----
  v16h A0, A1;
#pragma unroll
  for (int i = 0; i < 16; ++i) {
    int k = i + koff;
    A0[i] = (_Float16)((col == k)      ? 1.0f : 0.0f);
    A1[i] = (_Float16)((16 + col == k) ? 1.0f : 0.0f);
  }

  float acc = 0.0f;       // sum of log(Z) over steps done so far
  float lastLog = 0.0f;   // log(Z) of the most recent step
  const v8f zero = {0.f, 0.f, 0.f, 0.f, 0.f, 0.f, 0.f, 0.f};
  float* p = &sP[wid][0];
  const int rowadd = lo ? 0 : 8;   // C layout: lanes>=16 hold rows r+8

#define TT_W(aa, bb) \
  __builtin_amdgcn_wmma_f32_16x16x32_f16(false, aa, false, bb, (short)0, zero, false, false)
#define TT_MM(bb0, bb1)                       \
  {                                           \
    c00 = TT_W(A0, bb0);                      \
    c01 = TT_W(A0, bb1);                      \
    c10 = TT_W(A1, bb0);                      \
    c11 = TT_W(A1, bb1);                      \
  }

  for (int s = 0; s < 32; ++s) {
#pragma unroll
    for (int j = 0; j < 4; ++j) {
      int xw = (j == 0) ? xv.x : (j == 1) ? xv.y : (j == 2) ? xv.z : xv.w;
      int x = __builtin_amdgcn_readlane(xw, s);   // wave-uniform symbol

      v8f c00, c01, c10, c11;
      // Scalar 4-way branch: EXEC stays all-ones for WMMA, no VGPR-array indexing.
      if      (x == 0) TT_MM(Bt[0][0], Bt[0][1])
      else if (x == 1) TT_MM(Bt[1][0], Bt[1][1])
      else if (x == 2) TT_MM(Bt[2][0], Bt[2][1])
      else             TT_MM(Bt[3][0], Bt[3][1])

      // UNSCALED C -> LDS immediately: overlaps the max/log/rcp chain below.
#pragma unroll
      for (int r = 0; r < 8; ++r) {
        p[(r + rowadd) * TT_STR + col]           = c00[r];
        p[(r + rowadd) * TT_STR + 16 + col]      = c01[r];
        p[(16 + r + rowadd) * TT_STR + col]      = c10[r];
        p[(16 + r + rowadd) * TT_STR + 16 + col] = c11[r];
      }

      // inf-norm of the 32x32 product (per-lane tree folds to v_max3 with |.|)
      float mx = 0.0f;
#pragma unroll
      for (int i = 0; i < 8; ++i) {
        mx = fmaxf(mx, fabsf(c00[i]));
        mx = fmaxf(mx, fabsf(c01[i]));
        mx = fmaxf(mx, fabsf(c10[i]));
        mx = fmaxf(mx, fabsf(c11[i]));
      }
      mx = tt_wave_max(mx);           // pure VALU DPP butterfly
      lastLog = __logf(mx);
      acc += lastLog;
      float rz = __builtin_amdgcn_rcpf(mx);

      // LDS -> next A: scale by 1/Z here, pack f32 pairs to f16.
#pragma unroll
      for (int r = 0; r < 8; ++r) {
        int k = 2 * r + koff;
        float2 q0 = *(const float2*)&p[col * TT_STR + k];
        float2 q1 = *(const float2*)&p[(16 + col) * TT_STR + k];
        h2v h0 = tt_pack(q0.x * rz, q0.y * rz);
        h2v h1 = tt_pack(q1.x * rz, q1.y * rz);
        A0[2 * r] = h0[0]; A0[2 * r + 1] = h0[1];
        A1[2 * r] = h1[0]; A1[2 * r + 1] = h1[1];
      }
    }
  }
#undef TT_MM
#undef TT_W

  // LDS holds the last C *unscaled* (off by Z_last): publish acc - log(Z_last).
  if (lane == 0) sAcc[wid] = acc - lastLog;
  __syncthreads();

  // ---- Combine: v0 @ P0 @ P1 ... @ P7, then right boundary ----
  if (wid == 0) {
    float v  = lb[lane];
    float a2 = 0.0f;
#pragma unroll
    for (int w = 0; w < TT_NW; ++w) a2 += sAcc[w];
    float Z0 = tt_wave_max(fabsf(v));
    a2 += __logf(Z0);
    v *= __builtin_amdgcn_rcpf(Z0);

    for (int w = 0; w < TT_NW; ++w) {
      const float* Pw = &sP[w][0];
      float nv = 0.0f;
      for (int i = 0; i < 32; ++i) {
        float vi = __shfl(v, i, 32);               // broadcast v[i]
        nv = fmaf(vi, Pw[i * TT_STR + lane], nv);  // column `lane` of P_w
      }
      float Zw = tt_wave_max(fabsf(nv));
      a2 += __logf(Zw);
      v = nv * __builtin_amdgcn_rcpf(Zw);
    }

    float o = tt_wave_sum(v * rb[lane]);
    a2 += __logf(fabsf(o));
    if (lane == 0) out[b] = 2.0f * a2;
  }
}

extern "C" void kernel_launch(void* const* d_in, const int* in_sizes, int n_in,
                              void* d_out, int out_size, void* d_ws, size_t ws_size,
                              hipStream_t stream) {
  const int*   X    = (const int*)d_in[0];
  const float* core = (const float*)d_in[1];
  const float* lb   = (const float*)d_in[2];
  const float* rb   = (const float*)d_in[3];
  float* out = (float*)d_out;

  const int B = in_sizes[0] / TT_T;   // 256
  tt_logprob_kernel<<<B, 32 * TT_NW, 0, stream>>>(X, core, lb, rb, out);
  (void)n_in; (void)out_size; (void)d_ws; (void)ws_size;
}
